// ChatGLM3Attention_24721831756553
// MI455X (gfx1250) — matrix operand order repaired and checked by reading, not compiled, not run on hardware
//
#include <hip/hip_runtime.h>
#include <hip/hip_bf16.h>

// ---------------------------------------------------------------------------
// ChatGLM3 attention block for MI455X (gfx1250, wave32, WMMA bf16, TDM)
//   mixed = hs @ Wqkv^T + b   -> RoPE(q,k), fold 1/sqrt(128) into q
//   flash-attention (GQA 32q/2kv, hd=128, causal)
//   out = ctx @ Wdense^T
// GEMMs: v_wmma_f32_16x16x32_bf16 fed by Tensor-Data-Mover DMA into
// double-buffered LDS (tensor_load_to_lds + s_wait_tensorcnt).
// Block tile 64M x 256N (8 waves of 32M x 64N) -> 1.5 ds_load per WMMA and
// 2x better B-tile reuse than a 32-row block.
// ---------------------------------------------------------------------------

typedef __attribute__((ext_vector_type(4)))  __bf16 v4bf;
typedef __attribute__((ext_vector_type(8)))  __bf16 v8bf;
typedef __attribute__((ext_vector_type(16))) __bf16 v16bf;
typedef __attribute__((ext_vector_type(8)))  float  v8f;
typedef __attribute__((ext_vector_type(4)))  unsigned int v4u;
typedef __attribute__((ext_vector_type(8)))  int v8i;
typedef __attribute__((ext_vector_type(4)))  int v4i;

union BF16x16 { v16bf v; v8bf h[2]; };

#define HIDDEN  4096
#define NH      32
#define NKV     2
#define HD      128
#define QKV_OUT 4608          // NH*HD + 2*NKV*HD
#define ATT_SCALE 0.08838834764831845f   // 128^-0.5

static __device__ __forceinline__ v8f wmma_bf16(v16bf a, v16bf b, v8f c) {
  return __builtin_amdgcn_wmma_f32_16x16x32_bf16(
      /*neg_a=*/false, a, /*neg_b=*/false, b,
      /*c_mod=*/(short)0, c, /*reuse_a=*/false, /*reuse_b=*/false);
}

// ---------------------------------------------------------------------------
// TDM: DMA a 2-D bf16 tile (tile_h rows x tile_w elems, row stride in elems)
// from global memory into LDS at byte offset lds_off. Pad config inserts
// 4 DWORDs (8 bf16) after every 16 DWORDs (32 bf16) -> LDS row stride of 40
// bf16 elements, matching the padded fragment addressing below.
// D# bitfields per CDNA5 ISA.
// ---------------------------------------------------------------------------
static __device__ __forceinline__ void tdm_load_tile_2d(
    unsigned lds_off, const __bf16* gptr,
    unsigned tile_w, unsigned tile_h, unsigned long long row_stride_elems) {
  const unsigned long long ga = (unsigned long long)(const void*)gptr;
  v4u g0 = {
      1u,                                            // count=1 (valid, user)
      lds_off,                                       // lds_addr (bytes)
      (unsigned)ga,                                  // global_addr[31:0]
      (unsigned)((ga >> 32) & 0x01ffffffu) | (2u << 30)  // addr[56:32]|type=2
  };
  const unsigned d0 = tile_w;                        // tensor_dim0 == tile_w
  const unsigned d1 = tile_h;                        // tensor_dim1 == tile_h
  const unsigned long long s0 = row_stride_elems;    // tensor_dim0_stride
  v8i g1 = {
      // data_size=1 (2B) | pad_enable | pad_interval=3 (16 dw) | pad_amount=3 (4 dw)
      (int)((1u << 16) | (1u << 20) | (3u << 22) | (3u << 25)),
      (int)((d0 & 0xffffu) << 16),                           // td0[15:0]
      (int)(((d0 >> 16) & 0xffffu) | ((d1 & 0xffffu) << 16)),// td0[31:16]|td1[15:0]
      (int)(((d1 >> 16) & 0xffffu) | (tile_w << 16)),        // td1[31:16]|tile_dim0
      (int)(tile_h & 0xffffu),                               // tile_dim1 (tile_dim2=0)
      (int)(unsigned)(s0 & 0xffffffffu),                     // stride0[31:0]
      (int)(unsigned)((s0 >> 32) & 0xffffu),                 // stride0[47:32]
      0
  };
  v4i gz = {0, 0, 0, 0};                             // 2-D: groups 2/3 unused
#if defined(__clang_major__) && (__clang_major__ >= 23)
  v8i gz8 = {0, 0, 0, 0, 0, 0, 0, 0};
  __builtin_amdgcn_tensor_load_to_lds(g0, g1, gz, gz, gz8, 0);
#else
  __builtin_amdgcn_tensor_load_to_lds(g0, g1, gz, gz, 0);
#endif
}

// ---------------------------------------------------------------------------
// fp32 -> bf16 cast (vector x4 per thread)
// ---------------------------------------------------------------------------
__global__ __launch_bounds__(256)
void cast_f32_bf16(const float* __restrict__ in, __bf16* __restrict__ out,
                   size_t n) {
  size_t i = ((size_t)blockIdx.x * 256 + threadIdx.x) * 4;
  if (i >= n) return;
  float4 f = *(const float4*)(in + i);
  v4bf o;
  o[0] = (__bf16)f.x; o[1] = (__bf16)f.y; o[2] = (__bf16)f.z; o[3] = (__bf16)f.w;
  *(v4bf*)(out + i) = o;
}

// ---------------------------------------------------------------------------
// NT GEMM:  C[m,n] = sum_k A[m,k] * B[n,k]  (+ bias[n])
// A: MxK bf16 row-major, B: NxK bf16 row-major, C: MxN fp32.
// Block = 256 threads = 8 waves (2M x 4N); wave tile 32M x 64N (2x4 WMMA
// tiles, A-fragments reused across the 4 N-tiles). Block tile 64M x 256N,
// K-step 32. Tiles are DMA'd by the TDM into double-buffered padded LDS;
// wave 0 issues descriptors, completion gated by s_wait_tensorcnt before the
// block barrier; compute on buffer t overlaps the DMA filling buffer t+1.
// Assumes M%64==0, N%256==0, K%32==0 (true for all launches here).
// ---------------------------------------------------------------------------
__global__ __launch_bounds__(256)
void gemm_nt_bf16(const __bf16* __restrict__ A, const __bf16* __restrict__ Bw,
                  const float* __restrict__ bias, float* __restrict__ C,
                  int M, int N, int K) {
  __shared__ __bf16 As[2][64][40];     // TDM pad -> 40-elem rows, 16B aligned
  __shared__ __bf16 Bs[2][256][40];

  const int tid  = threadIdx.x;
  const int lane = tid & 31;
  const int wave = tid >> 5;
  const int half = lane >> 4;      // 0: lanes 0-15, 1: lanes 16-31
  const int l16  = lane & 15;
  const int m0 = blockIdx.y * 64;
  const int n0 = blockIdx.x * 256;
  const int wm = (wave >> 2) * 32; // 0 / 32
  const int wn = (wave & 3) * 64;  // 0 / 64 / 128 / 192

  const unsigned ldsA0 = (unsigned)(size_t)&As[0][0][0];
  const unsigned ldsA1 = (unsigned)(size_t)&As[1][0][0];
  const unsigned ldsB0 = (unsigned)(size_t)&Bs[0][0][0];
  const unsigned ldsB1 = (unsigned)(size_t)&Bs[1][0][0];

  v8f acc[2][4] = {};
  const int nT = K / 32;

  if (wave == 0) {   // prologue: fill buffer 0 (k0 = 0)
    tdm_load_tile_2d(ldsA0, A + (size_t)m0 * K, 32, 64, (unsigned long long)K);
    tdm_load_tile_2d(ldsB0, Bw + (size_t)n0 * K, 32, 256, (unsigned long long)K);
  }

  for (int t = 0; t < nT; ++t) {
    const int cur = t & 1;
    __builtin_amdgcn_s_wait_tensorcnt(0);   // buf[cur] ready (no-op for w1..7)
    __syncthreads();
    if (wave == 0 && (t + 1) < nT) {        // prefetch buf[nxt]
      const int k1 = (t + 1) * 32;
      tdm_load_tile_2d(cur ? ldsA0 : ldsA1,
                       A + (size_t)m0 * K + k1, 32, 64, (unsigned long long)K);
      tdm_load_tile_2d(cur ? ldsB0 : ldsB1,
                       Bw + (size_t)n0 * K + k1, 32, 256, (unsigned long long)K);
    }

    const __bf16* AsC = &As[cur][0][0];
    const __bf16* BsC = &Bs[cur][0][0];

    // A fragments (16x32 bf16, ISA layout: lane half selects K chunk 0/8)
    BF16x16 af[2];
#pragma unroll
    for (int i = 0; i < 2; ++i) {
      const int row = wm + i * 16 + l16;
      af[i].h[0] = *(const v8bf*)(AsC + row * 40 + half * 8);
      af[i].h[1] = *(const v8bf*)(AsC + row * 40 + half * 8 + 16);
    }
#pragma unroll
    for (int j = 0; j < 4; ++j) {
      // B fragment (32x16): column = l16, K chunks half*8 and half*8+16
      BF16x16 bf;
      const int col = wn + j * 16 + l16;
      bf.h[0] = *(const v8bf*)(BsC + col * 40 + half * 8);
      bf.h[1] = *(const v8bf*)(BsC + col * 40 + half * 8 + 16);
      acc[0][j] = wmma_bf16(af[0].v, bf.v, acc[0][j]);
      acc[1][j] = wmma_bf16(af[1].v, bf.v, acc[1][j]);
    }
  }

#pragma unroll
  for (int j = 0; j < 4; ++j) {
    const int col = n0 + wn + j * 16 + l16;
    const float bv = bias ? bias[col] : 0.0f;
#pragma unroll
    for (int i = 0; i < 2; ++i) {
#pragma unroll
      for (int v = 0; v < 8; ++v) {
        const int row = m0 + wm + i * 16 + v + half * 8;   // C/D layout
        C[(size_t)row * N + col] = acc[i][j][v] + bv;
      }
    }
  }
}

// ---------------------------------------------------------------------------
// Split QKV + GLM partial-rotary RoPE + scale(Q) + cast to bf16.
// One thread handles one (b,s,head,i), i in [0,64): i<32 rotates the
// interleaved pair (2i,2i+1) of the first 64 dims, i>=32 copies the tail pair.
// heads 0..31 -> Q (scaled), 32..33 -> K.
// ---------------------------------------------------------------------------
__global__ __launch_bounds__(256)
void rope_split(const float* __restrict__ mixed, const int* __restrict__ pos,
                __bf16* __restrict__ Qo, __bf16* __restrict__ Ko, int S) {
  const size_t idx = (size_t)blockIdx.x * 256 + threadIdx.x;
  const int    i    = (int)(idx & 63);
  const size_t row  = idx >> 6;              // (b*S+s)*34 + head
  const int    head = (int)(row % 34);
  const size_t bs   = row / 34;              // b*S + s
  const float  p    = (float)pos[bs];

  const float* src; __bf16* dst; float scale;
  if (head < NH) {
    src = mixed + bs * QKV_OUT + (size_t)head * HD;
    dst = Qo + (bs * NH + head) * HD;
    scale = ATT_SCALE;
  } else {
    const int kvh = head - NH;
    src = mixed + bs * QKV_OUT + NH * HD + (size_t)kvh * HD;
    dst = Ko + (bs * NKV + kvh) * HD;
    scale = 1.0f;
  }

  if (i < 32) {
    // inv_freq[i] = theta^-(2i/64)
    const float inv = __powf(10000.0f, -(float)(2 * i) * (1.0f / 64.0f));
    float sn, cs;
    __sincosf(p * inv, &sn, &cs);
    const float x1 = src[2 * i];
    const float x2 = src[2 * i + 1];
    dst[2 * i]     = (__bf16)((x1 * cs - x2 * sn) * scale);
    dst[2 * i + 1] = (__bf16)((x1 * sn + x2 * cs) * scale);
  } else {
    const int d = 64 + (i - 32) * 2;
    dst[d]     = (__bf16)(src[d] * scale);
    dst[d + 1] = (__bf16)(src[d + 1] * scale);
  }
}

// V slice: cast mixed[:, 4352:4608] -> bf16
__global__ __launch_bounds__(256)
void vcast(const float* __restrict__ mixed, __bf16* __restrict__ Vo) {
  const size_t idx = (size_t)blockIdx.x * 256 + threadIdx.x;   // B*S*512
  const size_t bs = idx >> 9;
  const int    c  = (int)(idx & 511);
  Vo[bs * 512 + c] = (__bf16)mixed[bs * QKV_OUT + NH * HD + NKV * HD + c];
}

// ---------------------------------------------------------------------------
// Causal flash attention, GQA (kv = h>>4). Block = 8 waves; wave owns a
// 16-query tile (full hd=128); block covers 128 queries of one (b,h).
// Key tiles of 32 staged in LDS; V staged transposed so P@V B-fragments are
// contiguous. Online softmax via half-wave shfl_xor reductions (rows of the
// WMMA C layout live across one 16-lane half per VGPR).
// ---------------------------------------------------------------------------
__global__ __launch_bounds__(256)
void attn_fwd(const __bf16* __restrict__ Q, const __bf16* __restrict__ Kc,
              const __bf16* __restrict__ Vc, __bf16* __restrict__ Ctx, int S) {
  __shared__ __bf16 Ks[32][136];        // [key][d], 272B rows (16B aligned)
  __shared__ __bf16 VsT[128][40];       // [d][key], transposed at stage time
  __shared__ __bf16 Ps[8][16][40];      // per-wave P relayout slab

  const int tid  = threadIdx.x;
  const int wave = tid >> 5;
  const int lane = tid & 31;
  const int half = lane >> 4;
  const int l16  = lane & 15;
  const int bh = blockIdx.y;
  const int b  = bh >> 5;
  const int h  = bh & 31;
  const int kv = h >> 4;                 // GQA: 16 q-heads per kv-head
  const int q0 = blockIdx.x * 128;
  const int qt = q0 + wave * 16;

  // --- load Q fragments for 4 K-steps of 32 over hd=128 (A layout) ---
  v16bf qf[4];
  {
    const size_t qbase = (((size_t)b * S + (qt + l16)) * NH + h) * HD;
#pragma unroll
    for (int s4 = 0; s4 < 4; ++s4) {
      BF16x16 t;
      t.h[0] = *(const v8bf*)(Q + qbase + s4 * 32 + half * 8);
      t.h[1] = *(const v8bf*)(Q + qbase + s4 * 32 + half * 8 + 16);
      qf[s4] = t.v;
    }
  }

  float mrow[8], lrow[8];
  v8f o[8] = {};
#pragma unroll
  for (int v = 0; v < 8; ++v) { mrow[v] = -1e30f; lrow[v] = 0.0f; }

  const int nkt  = q0 / 32 + 4;          // key tiles the block must stage
  const int myKt = (qt + 47) / 32;       // key tiles this wave computes

  for (int t = 0; t < nkt; ++t) {
    const int k0 = t * 32;
    __syncthreads();
    // stage K (row-major) and V (transposed); 512 8-elem chunks, 2/thread
    for (int u = tid; u < 512; u += 256) {
      const int r = u >> 4;              // key in tile 0..31
      const int c = (u & 15) * 8;        // d offset
      const size_t g = (((size_t)b * S + (k0 + r)) * NKV + kv) * HD + c;
      *(v8bf*)&Ks[r][c] = *(const v8bf*)(Kc + g);
      v8bf vv = *(const v8bf*)(Vc + g);
#pragma unroll
      for (int e = 0; e < 8; ++e) VsT[c + e][r] = vv[e];
    }
    __syncthreads();
    if (t >= myKt) continue;

    // --- S = Q @ K^T : two 16x16 key tiles, K-dim = hd in 4 steps ---
    v8f s0 = {}, s1 = {};
#pragma unroll
    for (int s4 = 0; s4 < 4; ++s4) {
      BF16x16 b0, b1;
      b0.h[0] = *(const v8bf*)&Ks[l16][s4 * 32 + half * 8];
      b0.h[1] = *(const v8bf*)&Ks[l16][s4 * 32 + half * 8 + 16];
      b1.h[0] = *(const v8bf*)&Ks[l16 + 16][s4 * 32 + half * 8];
      b1.h[1] = *(const v8bf*)&Ks[l16 + 16][s4 * 32 + half * 8 + 16];
      s0 = wmma_bf16(qf[s4], b0.v, s0);
      s1 = wmma_bf16(qf[s4], b1.v, s1);
    }

    // --- causal mask on diagonal tiles ---
    if (k0 + 31 > qt) {
#pragma unroll
      for (int v = 0; v < 8; ++v) {
        const int qm = qt + v + half * 8;
        if (k0 + l16 > qm)      s0[v] = -1e30f;
        if (k0 + 16 + l16 > qm) s1[v] = -1e30f;
      }
    }

    // --- online softmax (row = half-wave of one VGPR) ---
#pragma unroll
    for (int v = 0; v < 8; ++v) {
      float mx = fmaxf(s0[v], s1[v]);
#pragma unroll
      for (int d = 8; d >= 1; d >>= 1) mx = fmaxf(mx, __shfl_xor(mx, d, 32));
      const float mn    = fmaxf(mrow[v], mx);
      const float alpha = __expf(mrow[v] - mn);
      const float p0 = __expf(s0[v] - mn);
      const float p1 = __expf(s1[v] - mn);
      float sum = p0 + p1;
#pragma unroll
      for (int d = 8; d >= 1; d >>= 1) sum += __shfl_xor(sum, d, 32);
      lrow[v] = lrow[v] * alpha + sum;
      mrow[v] = mn;
      s0[v] = p0; s1[v] = p1;
#pragma unroll
      for (int c = 0; c < 8; ++c) o[c][v] *= alpha;
    }

    // --- re-layout P (C-layout f32) -> A-fragment bf16 via LDS slab ---
#pragma unroll
    for (int v = 0; v < 8; ++v) {
      const int m = v + half * 8;
      Ps[wave][m][l16]      = (__bf16)s0[v];
      Ps[wave][m][l16 + 16] = (__bf16)s1[v];
    }
    BF16x16 pf;   // wave-internal DS ops are in-order: safe without barrier
    pf.h[0] = *(const v8bf*)&Ps[wave][l16][half * 8];
    pf.h[1] = *(const v8bf*)&Ps[wave][l16][half * 8 + 16];

    // --- O += P @ V : 8 d-chunks of 16 ---
#pragma unroll
    for (int c = 0; c < 8; ++c) {
      BF16x16 vf;
      vf.h[0] = *(const v8bf*)&VsT[c * 16 + l16][half * 8];
      vf.h[1] = *(const v8bf*)&VsT[c * 16 + l16][half * 8 + 16];
      o[c] = wmma_bf16(pf.v, vf.v, o[c]);
    }
  }

  // --- epilogue: normalize, store ctx (B,S,NH*HD) as bf16 ---
#pragma unroll
  for (int v = 0; v < 8; ++v) {
    const float inv = 1.0f / lrow[v];
    const int q = qt + v + half * 8;
    const size_t base = (((size_t)b * S + q) * NH + h) * HD;
#pragma unroll
    for (int c = 0; c < 8; ++c) {
      Ctx[base + c * 16 + l16] = (__bf16)(o[c][v] * inv);
    }
  }
}

// ---------------------------------------------------------------------------
extern "C" void kernel_launch(void* const* d_in, const int* in_sizes, int n_in,
                              void* d_out, int out_size, void* d_ws,
                              size_t ws_size, hipStream_t stream) {
  (void)in_sizes; (void)n_in; (void)out_size; (void)ws_size;
  const int B = 2, S = 2048;
  const int M = B * S;                              // 4096 token rows

  const int*   positions = (const int*)d_in[0];
  const float* hidden    = (const float*)d_in[1];
  const float* w_qkv     = (const float*)d_in[2];
  const float* b_qkv     = (const float*)d_in[3];
  const float* w_dense   = (const float*)d_in[4];

  char* ws = (char*)d_ws;
  size_t off = 0;
  auto alloc = [&](size_t bytes) -> char* {
    char* p = ws + off;
    off += (bytes + 255) & ~(size_t)255;
    return p;
  };
  __bf16* hsB   = (__bf16*)alloc((size_t)M * HIDDEN * 2);        // 32 MB
  __bf16* wqkvB = (__bf16*)alloc((size_t)QKV_OUT * HIDDEN * 2);  // 36 MB
  __bf16* wdB   = (__bf16*)alloc((size_t)HIDDEN * HIDDEN * 2);   // 32 MB
  __bf16* qB    = (__bf16*)alloc((size_t)M * NH * HD * 2);       // 32 MB
  __bf16* kB    = (__bf16*)alloc((size_t)M * NKV * HD * 2);      //  2 MB
  __bf16* vB    = (__bf16*)alloc((size_t)M * NKV * HD * 2);      //  2 MB
  float*  mixed = (float*) alloc((size_t)M * QKV_OUT * 4);       // 72 MB
  __bf16* ctxB  = (__bf16*)mixed;   // reuse: mixed is dead after rope/vcast

  // 1) one-time bf16 conversions
  cast_f32_bf16<<<(M * HIDDEN) / 1024, 256, 0, stream>>>(
      hidden, hsB, (size_t)M * HIDDEN);
  cast_f32_bf16<<<(QKV_OUT * HIDDEN) / 1024, 256, 0, stream>>>(
      w_qkv, wqkvB, (size_t)QKV_OUT * HIDDEN);
  cast_f32_bf16<<<(HIDDEN * HIDDEN) / 1024, 256, 0, stream>>>(
      w_dense, wdB, (size_t)HIDDEN * HIDDEN);

  // 2) QKV projection (+bias): mixed = hs @ Wqkv^T + b
  gemm_nt_bf16<<<dim3(QKV_OUT / 256, M / 64), 256, 0, stream>>>(
      hsB, wqkvB, b_qkv, mixed, M, QKV_OUT, HIDDEN);

  // 3) RoPE + split + scale(Q) -> bf16 Q/K; cast V
  rope_split<<<((size_t)M * 34 * 64) / 256, 256, 0, stream>>>(
      mixed, positions, qB, kB, S);
  vcast<<<((size_t)M * NKV * HD) / 256, 256, 0, stream>>>(mixed, vB);

  // 4) causal GQA flash attention -> ctx bf16 (overwrites mixed region)
  attn_fwd<<<dim3(S / 128, B * NH), 256, 0, stream>>>(qB, kB, vB, ctxB, S);

  // 5) output projection: out = ctx @ Wdense^T (fp32 out)
  gemm_nt_bf16<<<dim3(HIDDEN / 256, M / 64), 256, 0, stream>>>(
      ctxB, wdB, nullptr, (float*)d_out, M, HIDDEN, HIDDEN);
}
